// Develop7_44392781971912
// MI455X (gfx1250) — matrix operand order repaired and checked by reading, not compiled, not run on hardware
//
#include <hip/hip_runtime.h>
#include <math.h>

// ---------------------------------------------------------------------------
// CDNA5 (gfx1250) wave32 WMMA types
// ---------------------------------------------------------------------------
typedef __attribute__((ext_vector_type(16))) __bf16 v16bf;
typedef __attribute__((ext_vector_type(8)))  float  v8f;

// ---------------------------------------------------------------------------
// embed: x (B,6,N) -> feats (B,N,32) = x^T @ W(6,32) + b ; coords = x^T[...,:3]
// ---------------------------------------------------------------------------
__global__ __launch_bounds__(256) void pt_embed(const float* __restrict__ x,
    const float* __restrict__ w, const float* __restrict__ b,
    float* __restrict__ feats, float* __restrict__ coords, int B, int N) {
  int t = blockIdx.x * 256 + threadIdx.x;
  if (t >= B * N) return;
  int bb = t / N, n = t - bb * N;
  float xi[6];
#pragma unroll
  for (int c = 0; c < 6; ++c) xi[c] = x[((size_t)bb * 6 + c) * N + n];
  coords[(size_t)t * 3 + 0] = xi[0];
  coords[(size_t)t * 3 + 1] = xi[1];
  coords[(size_t)t * 3 + 2] = xi[2];
#pragma unroll 4
  for (int o = 0; o < 32; ++o) {
    float s = b[o];
#pragma unroll
    for (int c = 0; c < 6; ++c) s += xi[c] * w[c * 32 + o];
    feats[(size_t)t * 32 + o] = s;
  }
}

// ---------------------------------------------------------------------------
// Farthest point sampling: one block per batch, sequential scan, block argmax.
// ---------------------------------------------------------------------------
__global__ __launch_bounds__(256) void pt_fps(const float* __restrict__ coords,
    int N, int npoint, int* __restrict__ cents) {
  int b = blockIdx.x;
  const float* c = coords + (size_t)b * N * 3;
  __shared__ float dist[2048];
  __shared__ float s_val[256];
  __shared__ int   s_idx[256];
  int t = threadIdx.x;
  for (int i = t; i < N; i += 256) dist[i] = 1e10f;
  __syncthreads();
  int farthest = 0;
  for (int it = 0; it < npoint; ++it) {
    if (t == 0) cents[(size_t)b * npoint + it] = farthest;
    float cx = c[(size_t)farthest * 3 + 0];
    float cy = c[(size_t)farthest * 3 + 1];
    float cz = c[(size_t)farthest * 3 + 2];
    float best = -1.0f; int bi = 0;
    for (int i = t; i < N; i += 256) {
      float dx = c[(size_t)i * 3 + 0] - cx;
      float dy = c[(size_t)i * 3 + 1] - cy;
      float dz = c[(size_t)i * 3 + 2] - cz;
      float d = dx * dx + dy * dy + dz * dz;
      float nd = fminf(dist[i], d);
      dist[i] = nd;
      if (nd > best) { best = nd; bi = i; }
    }
    s_val[t] = best; s_idx[t] = bi;
    __syncthreads();
    for (int o = 128; o > 0; o >>= 1) {
      if (t < o) {
        float v2 = s_val[t + o]; int i2 = s_idx[t + o];
        if (v2 > s_val[t] || (v2 == s_val[t] && i2 < s_idx[t])) { s_val[t] = v2; s_idx[t] = i2; }
      }
      __syncthreads();
    }
    farthest = s_idx[0];
    __syncthreads();
  }
}

// ---------------------------------------------------------------------------
// KNN: one block per (b, center). Iteratively extract 16 nearest (ties -> low idx).
// ---------------------------------------------------------------------------
__global__ __launch_bounds__(256) void pt_knn(const float* __restrict__ coords,
    const int* __restrict__ cents, int N, int P, int* __restrict__ knn) {
  int p = blockIdx.x, b = blockIdx.y;
  const float* c = coords + (size_t)b * N * 3;
  int ci = cents[(size_t)b * P + p];
  float sx = c[(size_t)ci * 3 + 0];
  float sy = c[(size_t)ci * 3 + 1];
  float sz = c[(size_t)ci * 3 + 2];
  __shared__ float dist[2048];
  __shared__ float s_val[256];
  __shared__ int   s_idx[256];
  int t = threadIdx.x;
  for (int i = t; i < N; i += 256) {
    float dx = c[(size_t)i * 3 + 0] - sx;
    float dy = c[(size_t)i * 3 + 1] - sy;
    float dz = c[(size_t)i * 3 + 2] - sz;
    dist[i] = dx * dx + dy * dy + dz * dz;
  }
  __syncthreads();
  for (int k = 0; k < 16; ++k) {
    float best = 3e38f; int bi = 0;
    for (int i = t; i < N; i += 256) {
      float d = dist[i];
      if (d < best) { best = d; bi = i; }
    }
    s_val[t] = best; s_idx[t] = bi;
    __syncthreads();
    for (int o = 128; o > 0; o >>= 1) {
      if (t < o) {
        float v2 = s_val[t + o]; int i2 = s_idx[t + o];
        if (v2 < s_val[t] || (v2 == s_val[t] && i2 < s_idx[t])) { s_val[t] = v2; s_idx[t] = i2; }
      }
      __syncthreads();
    }
    if (t == 0) { knn[((size_t)b * P + p) * 16 + k] = s_idx[0]; dist[s_idx[0]] = 3e38f; }
    __syncthreads();
  }
}

// ---------------------------------------------------------------------------
// Gather grouped coords/feats via knn indices. thread per (b,p,k).
// dim is always a multiple of 32 -> float4 copies are aligned.
// ---------------------------------------------------------------------------
__global__ __launch_bounds__(256) void pt_gather(const float* __restrict__ coords,
    const float* __restrict__ feats, const int* __restrict__ knn,
    float* __restrict__ g_coords, float* __restrict__ g_feats,
    int N, int P, int dim, int B) {
  size_t t = (size_t)blockIdx.x * 256 + threadIdx.x;
  size_t total = (size_t)B * P * 16;
  if (t >= total) return;
  int b = (int)(t / ((size_t)P * 16));
  int idx = knn[t];
  const float* cs = coords + ((size_t)b * N + idx) * 3;
  g_coords[t * 3 + 0] = cs[0];
  g_coords[t * 3 + 1] = cs[1];
  g_coords[t * 3 + 2] = cs[2];
  const float4* fs = (const float4*)(feats + ((size_t)b * N + idx) * dim);
  float4* fd = (float4*)(g_feats + t * (size_t)dim);
  int n4 = dim >> 2;
  for (int d = 0; d < n4; ++d) fd[d] = fs[d];
}

// ---------------------------------------------------------------------------
// Positional encoding: rel = c_i - c_j -> LN(3) -> 3x64 relu -> 64xH.
// thread per (b,p,i,j). Output pos (B,P,16,16,H).
// ---------------------------------------------------------------------------
__global__ __launch_bounds__(256) void pt_pos(const float* __restrict__ g_coords,
    const float* __restrict__ ln_g, const float* __restrict__ ln_b,
    const float* __restrict__ w1, const float* __restrict__ b1,
    const float* __restrict__ w2, const float* __restrict__ b2,
    float* __restrict__ pos, int P, int H, int B) {
  size_t t = (size_t)blockIdx.x * 256 + threadIdx.x;
  size_t total = (size_t)B * P * 256;
  if (t >= total) return;
  int j = (int)(t & 15);
  int i = (int)((t >> 4) & 15);
  size_t bp = t >> 8;
  const float* cb = g_coords + bp * 16 * 3;
  float r0 = cb[i * 3 + 0] - cb[j * 3 + 0];
  float r1 = cb[i * 3 + 1] - cb[j * 3 + 1];
  float r2 = cb[i * 3 + 2] - cb[j * 3 + 2];
  float m = (r0 + r1 + r2) * (1.0f / 3.0f);
  float d0 = r0 - m, d1 = r1 - m, d2 = r2 - m;
  float var = (d0 * d0 + d1 * d1 + d2 * d2) * (1.0f / 3.0f);
  float inv = rsqrtf(var + 1e-5f);
  float n0 = d0 * inv * ln_g[0] + ln_b[0];
  float n1 = d1 * inv * ln_g[1] + ln_b[1];
  float n2 = d2 * inv * ln_g[2] + ln_b[2];
  float h[64];
#pragma unroll 8
  for (int o = 0; o < 64; ++o) {
    float v = n0 * w1[o] + n1 * w1[64 + o] + n2 * w1[128 + o] + b1[o];
    h[o] = fmaxf(v, 0.0f);
  }
  float* pd = pos + t * (size_t)H;
  for (int hh = 0; hh < H; ++hh) {
    float acc = b2[hh];
    for (int o = 0; o < 64; ++o) acc += h[o] * w2[o * H + hh];
    pd[hh] = acc;
  }
}

// ---------------------------------------------------------------------------
// LayerNorm over last dim, one wave32 per row.
// ---------------------------------------------------------------------------
__global__ __launch_bounds__(256) void pt_layernorm(const float* __restrict__ x,
    const float* __restrict__ g, const float* __restrict__ b,
    float* __restrict__ y, size_t rows, int dim) {
  size_t row = (size_t)blockIdx.x * 8 + (threadIdx.x >> 5);
  int lane = threadIdx.x & 31;
  if (row >= rows) return;
  const float* xr = x + row * (size_t)dim;
  float s = 0.f;
  for (int i = lane; i < dim; i += 32) s += xr[i];
#pragma unroll
  for (int o = 16; o > 0; o >>= 1) s += __shfl_xor(s, o, 32);
  float mean = s / dim;
  float vs = 0.f;
  for (int i = lane; i < dim; i += 32) { float d = xr[i] - mean; vs += d * d; }
#pragma unroll
  for (int o = 16; o > 0; o >>= 1) vs += __shfl_xor(vs, o, 32);
  float inv = rsqrtf(vs / dim + 1e-5f);
  float* yr = y + row * (size_t)dim;
  for (int i = lane; i < dim; i += 32) yr[i] = (xr[i] - mean) * inv * g[i] + b[i];
}

// ---------------------------------------------------------------------------
// Attention over K=16 neighborhood. thread per (b,p,head,row).
// qkv rows: (B*P*16, 3*inner) [q|k|v]; pos (B,P,16,16,H); out (B*P*16, inner).
// ---------------------------------------------------------------------------
__global__ __launch_bounds__(256) void pt_attn(const float* __restrict__ qkv,
    const float* __restrict__ pos, float* __restrict__ out,
    int P, int H, int dh, int B) {
  size_t t = (size_t)blockIdx.x * 256 + threadIdx.x;
  size_t total = (size_t)B * P * H * 16;
  if (t >= total) return;
  int i = (int)(t & 15);
  size_t r = t >> 4;
  int h = (int)(r % H);
  size_t bp = r / H;
  int inner = H * dh;
  size_t stride = 3 * (size_t)inner;
  const float* base = qkv + bp * 16 * stride;
  const float* qrow = base + (size_t)i * stride + h * dh;
  const float* pb = pos + (bp * 16 + i) * 16 * (size_t)H + h;
  float scale = rsqrtf((float)dh);
  float dots[16];
  float mx = -3e38f;
  for (int j = 0; j < 16; ++j) {
    const float* krow = base + (size_t)j * stride + inner + h * dh;
    float d = 0.f;
    for (int d0 = 0; d0 < dh; ++d0) d += qrow[d0] * krow[d0];
    d = (d + pb[(size_t)j * H]) * scale;
    dots[j] = d;
    mx = fmaxf(mx, d);
  }
  float se = 0.f;
  for (int j = 0; j < 16; ++j) { dots[j] = expf(dots[j] - mx); se += dots[j]; }
  float rinv = 1.0f / se;
  float* orow = out + (bp * 16 + i) * (size_t)inner + h * dh;
  for (int d0 = 0; d0 < dh; ++d0) {
    float acc = 0.f;
    for (int j = 0; j < 16; ++j)
      acc += dots[j] * base[(size_t)j * stride + 2 * (size_t)inner + h * dh + d0];
    orow[d0] = acc * rinv;
  }
}

// ---------------------------------------------------------------------------
// Weight transpose + f32->bf16 convert: Wt[n*K + k] = (bf16) W[k*N + n].
// ---------------------------------------------------------------------------
__global__ __launch_bounds__(256) void pt_cvtw(const float* __restrict__ W,
    __bf16* __restrict__ Wt, int K, int N) {
  int t = blockIdx.x * 256 + threadIdx.x;
  if (t >= K * N) return;
  int n = t / K, k = t - n * K;
  Wt[t] = (__bf16)W[(size_t)k * N + n];
}

// ---------------------------------------------------------------------------
// WMMA bf16 GEMM: C[M,N] = act(A[M,K] @ W[K,N] + bias [+ residual]).
// Requirements (guaranteed by all call sites): K % 32 == 0, M % 16 == 0.
// Wt is W transposed to [N][K] in bf16 (pt_cvtw).
// One wave32 per 16x32 output tile (two WMMA accumulators share the A frag).
// A frag (16-bit A 16x32, contiguous k order chosen here, same as B): lane l (m=l&15, half=l>>4):
//   elem i -> k = half*16+i
//   => two aligned contiguous v8f loads per chunk.
// B frag (32x16): lane l (n=l&15): elem i -> k = half*16+i
//   => one aligned contiguous v16bf load from Wt per tile per chunk.
// C/D: vgpr r, lane l -> (m = r + half*8, n = l&15)
// ACT: 0 none, 1 relu, 2 exact gelu (erf)
// ---------------------------------------------------------------------------
template <int ACT>
__global__ __launch_bounds__(256) void pt_gemm(const float* __restrict__ A,
    const __bf16* __restrict__ Wt, const float* __restrict__ bias,
    const float* __restrict__ residual, float* __restrict__ C,
    int M, int K, int N) {
  int lane = threadIdx.x & 31;
  int wave = threadIdx.x >> 5;
  int mt = blockIdx.y * 8 + wave;
  int m0 = mt * 16;
  if (m0 >= M) return;
  int n0 = blockIdx.x * 32;
  int half = lane >> 4;
  int lm = lane & 15;
  int arow = m0 + lm;                       // always < M (M % 16 == 0)
  int ncol0 = n0 + lm;
  int ncol1 = n0 + 16 + lm;
  int nc0 = ncol0 < N ? ncol0 : N - 1;      // clamp for loads; stores guarded
  int nc1 = ncol1 < N ? ncol1 : N - 1;

  const float*  pa  = A  + (size_t)arow * K + half * 16;
  const __bf16* pb0 = Wt + (size_t)nc0 * K + half * 16;
  const __bf16* pb1 = Wt + (size_t)nc1 * K + half * 16;

  v8f acc0 = {};
  v8f acc1 = {};
  for (int k0 = 0; k0 < K; k0 += 32) {
    v8f af0 = *(const v8f*)(pa + k0);        // k = k0 + half*16 + [0..7]
    v8f af1 = *(const v8f*)(pa + k0 + 8);    // k = k0 + half*16 + 8 + [0..7]
    v16bf a;
#pragma unroll
    for (int ii = 0; ii < 8; ++ii) {
      a[ii]     = (__bf16)af0[ii];
      a[ii + 8] = (__bf16)af1[ii];
    }
    v16bf b0 = *(const v16bf*)(pb0 + k0);
    v16bf b1 = *(const v16bf*)(pb1 + k0);
    acc0 = __builtin_amdgcn_wmma_f32_16x16x32_bf16(false, a, false, b0,
                                                   (short)0, acc0, false, false);
    acc1 = __builtin_amdgcn_wmma_f32_16x16x32_bf16(false, a, false, b1,
                                                   (short)0, acc1, false, false);
  }

  float bv0 = bias ? bias[nc0] : 0.0f;
  float bv1 = bias ? bias[nc1] : 0.0f;
  if (ncol0 < N) {
#pragma unroll
    for (int r = 0; r < 8; ++r) {
      int m = m0 + half * 8 + r;
      float v = acc0[r] + bv0;
      if (residual) v += residual[(size_t)m * N + ncol0];
      if (ACT == 1) v = fmaxf(v, 0.0f);
      else if (ACT == 2) v = 0.5f * v * (1.0f + erff(v * 0.70710678118654752f));
      C[(size_t)m * N + ncol0] = v;
    }
  }
  if (ncol1 < N) {
#pragma unroll
    for (int r = 0; r < 8; ++r) {
      int m = m0 + half * 8 + r;
      float v = acc1[r] + bv1;
      if (residual) v += residual[(size_t)m * N + ncol1];
      if (ACT == 1) v = fmaxf(v, 0.0f);
      else if (ACT == 2) v = 0.5f * v * (1.0f + erff(v * 0.70710678118654752f));
      C[(size_t)m * N + ncol1] = v;
    }
  }
}

// ---------------------------------------------------------------------------
// Extract first neighbor (k=0): new coords and sampled feats.
// ---------------------------------------------------------------------------
__global__ __launch_bounds__(256) void pt_first(const float* __restrict__ g_coords,
    const float* __restrict__ feats, float* __restrict__ coords_out,
    float* __restrict__ sp, int P, int dim, int B) {
  size_t t = (size_t)blockIdx.x * 256 + threadIdx.x;
  if (t >= (size_t)B * P) return;
  size_t row = t * 16;
  coords_out[t * 3 + 0] = g_coords[row * 3 + 0];
  coords_out[t * 3 + 1] = g_coords[row * 3 + 1];
  coords_out[t * 3 + 2] = g_coords[row * 3 + 2];
  const float4* fs = (const float4*)(feats + row * (size_t)dim);
  float4* sd = (float4*)(sp + t * (size_t)dim);
  int n4 = dim >> 2;
  for (int d = 0; d < n4; ++d) sd[d] = fs[d];
}

// ---------------------------------------------------------------------------
// Mean over points axis: (B,P,D) -> (B,D)
// ---------------------------------------------------------------------------
__global__ __launch_bounds__(256) void pt_mean(const float* __restrict__ x,
    float* __restrict__ y, int B, int P, int D) {
  int t = blockIdx.x * 256 + threadIdx.x;
  if (t >= B * D) return;
  int b = t / D, d = t - b * D;
  float s = 0.f;
  for (int p = 0; p < P; ++p) s += x[((size_t)b * P + p) * D + d];
  y[t] = s / (float)P;
}

// ---------------------------------------------------------------------------
// host-side GEMM dispatch helper (converts weights, then runs WMMA GEMM)
// ---------------------------------------------------------------------------
static void launch_gemm(const float* A, const float* W, const float* bias,
                        const float* residual, float* C, size_t M, int K, int N,
                        int act, __bf16* wt, hipStream_t stream) {
  int total = K * N;
  pt_cvtw<<<(total + 255) / 256, 256, 0, stream>>>(W, wt, K, N);
  dim3 grid((unsigned)((N + 31) / 32), (unsigned)((M + 127) / 128));
  if (act == 0)
    pt_gemm<0><<<grid, 256, 0, stream>>>(A, wt, bias, residual, C, (int)M, K, N);
  else if (act == 1)
    pt_gemm<1><<<grid, 256, 0, stream>>>(A, wt, bias, residual, C, (int)M, K, N);
  else
    pt_gemm<2><<<grid, 256, 0, stream>>>(A, wt, bias, residual, C, (int)M, K, N);
}

// ---------------------------------------------------------------------------
extern "C" void kernel_launch(void* const* d_in, const int* in_sizes, int n_in,
                              void* d_out, int out_size, void* d_ws, size_t ws_size,
                              hipStream_t stream) {
  (void)in_sizes; (void)n_in; (void)out_size; (void)ws_size;
  int cur = 0;
  auto nextp = [&]() { return (const float*)d_in[cur++]; };

  const float* x       = nextp();
  const float* embed_w = nextp();
  const float* embed_b = nextp();

  const int Bc = 32, Nc = 2048;
  char* ws = (char*)d_ws;
  size_t off = 0;
  auto alloc = [&](size_t bytes) -> void* {
    void* p = ws + off;
    off += (bytes + 255) & ~(size_t)255;
    return p;
  };

  // persistent ping-pong buffers across stages
  float* coordsA = (float*)alloc((size_t)Bc * Nc * 3 * sizeof(float));
  float* coordsB = (float*)alloc((size_t)Bc * Nc * 3 * sizeof(float));
  float* featsA  = (float*)alloc((size_t)Bc * Nc * 32 * sizeof(float));
  float* featsB  = (float*)alloc((size_t)Bc * Nc * 32 * sizeof(float));
  size_t scratch_base = off;

  pt_embed<<<(Bc * Nc + 255) / 256, 256, 0, stream>>>(x, embed_w, embed_b,
                                                      featsA, coordsA, Bc, Nc);

  float* coords_in  = coordsA;
  float* feats_in   = featsA;
  float* coords_out = coordsB;
  float* feats_out  = featsB;
  int Pin = Nc;

  const int nb_[4]  = {1, 2, 1, 1};
  const int Pn_[4]  = {512, 128, 32, 8};
  const int dim_[4] = {32, 64, 128, 256};
  const int H_[4]   = {8, 16, 16, 32};
  const int dh_[4]  = {16, 16, 32, 32};

  for (int s = 0; s < 4; ++s) {
    off = scratch_base;  // reuse scratch each stage (serial stream => safe)
    int P = Pn_[s], dim = dim_[s], H = H_[s], dh = dh_[s];
    int inner = H * dh;
    size_t rows = (size_t)Bc * P * 16;

    int*    cents    = (int*)alloc((size_t)Bc * P * sizeof(int));
    int*    knn      = (int*)alloc((size_t)Bc * P * 16 * sizeof(int));
    float*  g_coords = (float*)alloc(rows * 3 * sizeof(float));
    float*  fcur     = (float*)alloc(rows * dim * sizeof(float));
    float*  fnext    = (float*)alloc(rows * dim * sizeof(float));
    float*  lnbuf    = (float*)alloc(rows * dim * sizeof(float));
    float*  qkv      = (float*)alloc(rows * 3 * (size_t)inner * sizeof(float));
    float*  posb     = (float*)alloc(rows * 16 * (size_t)H * sizeof(float));
    float*  attnout  = (float*)alloc(rows * (size_t)inner * sizeof(float));
    float*  attb     = (float*)alloc(rows * dim * sizeof(float));
    float*  hbuf     = (float*)alloc(rows * dim * sizeof(float));
    float*  sp       = (float*)alloc((size_t)Bc * P * dim * sizeof(float));
    __bf16* wtbuf    = (__bf16*)alloc((size_t)dim * 3 * inner * sizeof(__bf16));

    pt_fps<<<Bc, 256, 0, stream>>>(coords_in, Pin, P, cents);
    pt_knn<<<dim3(P, Bc), 256, 0, stream>>>(coords_in, cents, Pin, P, knn);
    pt_gather<<<(unsigned)((rows + 255) / 256), 256, 0, stream>>>(
        coords_in, feats_in, knn, g_coords, fcur, Pin, P, dim, Bc);

    for (int blk = 0; blk < nb_[s]; ++blk) {
      const float* ln1_g = nextp(); const float* ln1_b = nextp();
      const float* ln2_g = nextp(); const float* ln2_b = nextp();
      const float* wqkv  = nextp();
      const float* wo    = nextp(); const float* bo    = nextp();
      const float* fw1   = nextp(); const float* fb1   = nextp();
      const float* fw2   = nextp(); const float* fb2   = nextp();
      const float* pln_g = nextp(); const float* pln_b = nextp();
      const float* pw1   = nextp(); const float* pb1   = nextp();
      const float* pw2   = nextp(); const float* pb2   = nextp();

      pt_pos<<<(unsigned)((rows * 16 + 255) / 256), 256, 0, stream>>>(
          g_coords, pln_g, pln_b, pw1, pb1, pw2, pb2, posb, P, H, Bc);

      pt_layernorm<<<(unsigned)((rows + 7) / 8), 256, 0, stream>>>(
          fcur, ln1_g, ln1_b, lnbuf, rows, dim);

      // qkv = ln1(x) @ wqkv           (M=rows, K=dim, N=3*inner)
      launch_gemm(lnbuf, wqkv, nullptr, nullptr, qkv, rows, dim, 3 * inner, 0,
                  wtbuf, stream);

      pt_attn<<<(unsigned)((rows * H + 255) / 256), 256, 0, stream>>>(
          qkv, posb, attnout, P, H, dh, Bc);

      // att = attnout @ wo + bo + feats (residual)
      launch_gemm(attnout, wo, bo, fcur, attb, rows, inner, dim, 0, wtbuf, stream);

      pt_layernorm<<<(unsigned)((rows + 7) / 8), 256, 0, stream>>>(
          attb, ln2_g, ln2_b, lnbuf, rows, dim);

      // h = gelu(ln2(att) @ fw1 + fb1)
      launch_gemm(lnbuf, fw1, fb1, nullptr, hbuf, rows, dim, dim, 2, wtbuf, stream);
      // out = h @ fw2 + fb2 + att (residual)
      launch_gemm(hbuf, fw2, fb2, attb, fnext, rows, dim, dim, 0, wtbuf, stream);

      float* tmp = fcur; fcur = fnext; fnext = tmp;
    }

    const float* down_w = nextp();
    const float* down_b = nextp();

    pt_first<<<(unsigned)(((size_t)Bc * P + 255) / 256), 256, 0, stream>>>(
        g_coords, fcur, coords_out, sp, P, dim, Bc);

    // feats_out = relu(sp @ down_w + down_b)   (M=B*P, K=dim, N=2*dim)
    launch_gemm(sp, down_w, down_b, nullptr, feats_out,
                (size_t)Bc * P, dim, 2 * dim, 1, wtbuf, stream);

    float* tc = coords_in; coords_in = coords_out; coords_out = tc;
    float* tf = feats_in;  feats_in  = feats_out;  feats_out  = tf;
    Pin = P;
  }

  const float* cw = nextp();
  const float* cb = nextp();

  float*  pooled = (float*)alloc((size_t)Bc * 512 * sizeof(float));
  __bf16* wtcls  = (__bf16*)alloc((size_t)512 * 40 * sizeof(__bf16));
  pt_mean<<<(Bc * 512 + 255) / 256, 256, 0, stream>>>(feats_in, pooled, Bc, 8, 512);

  // logits = pooled @ classify_w + classify_b  (M=32, K=512, N=40)
  launch_gemm(pooled, cw, cb, nullptr, (float*)d_out, Bc, 512, 40, 0, wtcls, stream);
}
